// ESNForecaster_41781441855561
// MI455X (gfx1250) — matrix operand order repaired and checked by reading, not compiled, not run on hardware
//
#include <hip/hip_runtime.h>

// ============================================================================
// ESN forward for MI455X (gfx1250, wave32).
//
// r_{t+1} = 0.4 r_t + 0.6 tanh(r_t A^T + B x_t + b),  1024 steps, then C r.
//
// Design (see analysis): split-bf16 (hi+lo) representation of A and r gives
// near-fp32 accuracy at bf16 WMMA rate; A_hi/A_lo (64MB) live in workspace and
// stay resident in the 192MB L2 across all 1024 steps. One persistent kernel
// with a device-wide atomic barrier per step avoids 1024 graph-node launches.
// 64 WGs x 128 threads (4 waves); each wave owns one 16x16 (batch x column)
// WMMA tile, K=4096 in chunks of 32, dual accumulators. r is staged to LDS as
// bf16 hi/lo each step (padded row stride 4104 -> conflict-free ds_load_b128).
//
// Workspace: A_hi 32MB | A_lo 32MB | rbuf0 256KB | rbuf1 256KB | counter.
// ============================================================================

#define NRR     4096
#define NBATCH  16
#define NFEAT   12
#define SEQLEN  1024
#define NOUT    8
#define NWG     64
#define TPB     128
#define LSTR    4104   // padded LDS row stride (bf16 elements): 4096 + 8

typedef __attribute__((ext_vector_type(16))) __bf16 v16bf;
typedef __attribute__((ext_vector_type(8)))  float  v8f;

struct __attribute__((aligned(16))) U4  { unsigned int   a, b, c, d; };
struct __attribute__((aligned(8)))  US4 { unsigned short a, b, c, d; };
union  Frag { v16bf v; U4 q[2]; };

__device__ __forceinline__ unsigned short bf16_rne(float f) {
    unsigned u = __float_as_uint(f);
    unsigned r = 0x7FFFu + ((u >> 16) & 1u);
    return (unsigned short)((u + r) >> 16);
}
__device__ __forceinline__ float bf16_f32(unsigned short h) {
    return __uint_as_float(((unsigned)h) << 16);
}
__device__ __forceinline__ void split2(float x, unsigned short& h, unsigned short& l) {
    h = bf16_rne(x);
    l = bf16_rne(x - bf16_f32(h));
}

__device__ __forceinline__ v8f wmma_bf16(Frag a, Frag b, v8f c) {
    // v_wmma_f32_16x16x32_bf16: D = A(16x32) * B(32x16) + C
    return __builtin_amdgcn_wmma_f32_16x16x32_bf16(
        false, a.v, false, b.v, (short)0, c, false, false);
}

// One K=32 chunk: 3 WMMAs implementing (r_hi+r_lo)*(A_hi+A_lo), dropping lo*lo.
__device__ __forceinline__ void gemm_chunk(const unsigned short* rh,
                                           const unsigned short* rl,
                                           const unsigned short* ah,
                                           const unsigned short* al,
                                           int k0, int hlf, v8f& acc) {
    const int ka = k0 + hlf * 16;   // A-operand: lane half selects K half
    const int kb = k0 + hlf * 16;   // B-operand: lane half selects K half
    Frag fah, fal, fbh, fbl;
    fah.q[0] = *(const U4*)(rh + ka);        // K = ka..ka+7
    fah.q[1] = *(const U4*)(rh + ka + 8);    // K = ka+8..ka+15
    fal.q[0] = *(const U4*)(rl + ka);
    fal.q[1] = *(const U4*)(rl + ka + 8);
    fbh.q[0] = *(const U4*)(ah + kb);        // K = kb..kb+7   (row-major A)
    fbh.q[1] = *(const U4*)(ah + kb + 8);    // K = kb+8..kb+15
    fbl.q[0] = *(const U4*)(al + kb);
    fbl.q[1] = *(const U4*)(al + kb + 8);
    acc = wmma_bf16(fah, fbh, acc);
    acc = wmma_bf16(fah, fbl, acc);
    acc = wmma_bf16(fal, fbh, acc);
}

// ---------------------------------------------------------------------------
// Pre-pass 1: split fp32 A into bf16 hi/lo pair (once per launch).
// ---------------------------------------------------------------------------
__global__ void __launch_bounds__(256) esn_split_A(const float4* __restrict__ A4,
                                                   uint2* __restrict__ hi2,
                                                   uint2* __restrict__ lo2) {
    int i = blockIdx.x * 256 + threadIdx.x;       // 4,194,304 float4s total
    if (i >= (NRR * NRR) / 4) return;
    float4 v = A4[i];
    unsigned short h0, h1, h2, h3, l0, l1, l2, l3;
    split2(v.x, h0, l0); split2(v.y, h1, l1);
    split2(v.z, h2, l2); split2(v.w, h3, l3);
    hi2[i] = make_uint2(((unsigned)h1 << 16) | h0, ((unsigned)h3 << 16) | h2);
    lo2[i] = make_uint2(((unsigned)l1 << 16) | l0, ((unsigned)l3 << 16) | l2);
}

// ---------------------------------------------------------------------------
// Pre-pass 2: state init + barrier counter reset (graph-replay safe).
// ---------------------------------------------------------------------------
__global__ void __launch_bounds__(256) esn_init(const float* __restrict__ r0,
                                                float* __restrict__ rbuf0,
                                                unsigned* __restrict__ counter) {
    int i = blockIdx.x * 256 + threadIdx.x;
    if (i < NBATCH * NRR) rbuf0[i] = r0[i];
    if (i == 0) *counter = 0u;
}

// ---------------------------------------------------------------------------
// Persistent recurrence kernel.
// ---------------------------------------------------------------------------
__global__ void __launch_bounds__(TPB, 1) esn_recurrence(
    const unsigned short* __restrict__ Ahi,   // (NRR, NRR) bf16
    const unsigned short* __restrict__ Alo,   // (NRR, NRR) bf16
    float* __restrict__ rbuf0,                // (NBATCH, NRR)
    float* __restrict__ rbuf1,                // (NBATCH, NRR)
    unsigned* __restrict__ counter,           // grid barrier (monotonic)
    const float* __restrict__ xin,            // (NBATCH, SEQLEN, NFEAT)
    const float* __restrict__ Bmat,           // (NRR, NFEAT)
    const float* __restrict__ bias,           // (NRR)
    const float* __restrict__ Cw,             // (NOUT, NRR)
    float* __restrict__ out)                  // (NBATCH, NOUT)
{
    extern __shared__ unsigned char smem_raw[];
    unsigned short* s_hi = (unsigned short*)smem_raw;          // [16][LSTR] bf16
    unsigned short* s_lo = s_hi + NBATCH * LSTR;               // [16][LSTR] bf16
    float* s_x  = (float*)(s_lo + NBATCH * LSTR);              // [16*12] x_t
    float* s_Bc = s_x + NBATCH * NFEAT;                        // [64*12] B cols
    float* s_bs = s_Bc + 64 * NFEAT;                           // [64] bias

    const int tid  = threadIdx.x;
    const int wg   = blockIdx.x;
    const int wave = tid >> 5;
    const int lane = tid & 31;
    const int hlf  = lane >> 4;        // lane half (K sub-block select)
    const int l16  = lane & 15;        // row (A-op) / column (B-op) within tile
    const int nwg0 = wg * 64;          // first reservoir column owned by WG
    const int n0   = nwg0 + wave * 16; // this wave's 16-column tile base

    // Per-WG constants: input-projection columns of B and bias.
    for (int i = tid; i < 64 * NFEAT; i += TPB) {
        int nl = i / NFEAT, f = i - nl * NFEAT;
        s_Bc[i] = Bmat[(size_t)(nwg0 + nl) * NFEAT + f];
    }
    if (tid < 64) s_bs[tid] = bias[nwg0 + tid];
    // visibility covered by the first __syncthreads in the step loop

    const unsigned short* arow_hi = Ahi + (size_t)(n0 + l16) * NRR;
    const unsigned short* arow_lo = Alo + (size_t)(n0 + l16) * NRR;
    const unsigned short* rrow_hi = s_hi + l16 * LSTR;
    const unsigned short* rrow_lo = s_lo + l16 * LSTR;

    for (int t = 0; t < SEQLEN; ++t) {
        const float* rc = (t & 1) ? rbuf1 : rbuf0;
        float*       rn = (t & 1) ? rbuf0 : rbuf1;

        // ---- stage x_t and r_t (fp32 -> bf16 hi/lo) into LDS ----
        if (tid < NBATCH * NFEAT) {
            int bq = tid / NFEAT, f = tid - bq * NFEAT;
            s_x[tid] = xin[((size_t)bq * SEQLEN + t) * NFEAT + f];
        }
        const float4* rc4 = (const float4*)rc;
        for (int i = tid; i < (NBATCH * NRR) / 4; i += TPB) {
            int m  = i >> 10;          // 1024 float4s per reservoir row
            int c4 = i & 1023;
            float4 v = rc4[i];
            unsigned short h0, h1, h2, h3, l0, l1, l2, l3;
            split2(v.x, h0, l0); split2(v.y, h1, l1);
            split2(v.z, h2, l2); split2(v.w, h3, l3);
            int e = m * LSTR + c4 * 4;
            US4 sh = {h0, h1, h2, h3};
            US4 sl = {l0, l1, l2, l3};
            *(US4*)(s_hi + e) = sh;
            *(US4*)(s_lo + e) = sl;
        }
        __syncthreads();

        // ---- 16x16 tile GEMM over K=4096, dual accumulators ----
        v8f acc0 = {}; v8f acc1 = {};
        for (int k0 = 0; k0 < NRR; k0 += 64) {
            gemm_chunk(rrow_hi, rrow_lo, arow_hi, arow_lo, k0,      hlf, acc0);
            gemm_chunk(rrow_hi, rrow_lo, arow_hi, arow_lo, k0 + 32, hlf, acc1);
        }

        // ---- z = GEMM + B x_t + b ; leak + tanh ; write next state ----
        const int nl = wave * 16 + l16;  // column within WG [0,64)
        const int ng = nwg0 + nl;        // global column
        #pragma unroll
        for (int j = 0; j < 8; ++j) {
            int m = j + 8 * hlf;         // D-matrix layout: M = j + 8*(lane>=16)
            float z = acc0[j] + acc1[j] + s_bs[nl];
            #pragma unroll
            for (int f = 0; f < NFEAT; ++f)
                z += s_x[m * NFEAT + f] * s_Bc[nl * NFEAT + f];
            float rold = rc[(size_t)m * NRR + ng];
            rn[(size_t)m * NRR + ng] = 0.4f * rold + 0.6f * tanhf(z);
        }

        // ---- device-wide barrier (monotonic counter, no reset needed) ----
        __syncthreads();
        if (tid == 0) {
            __builtin_amdgcn_fence(__ATOMIC_RELEASE, "agent");
            __hip_atomic_fetch_add(counter, 1u, __ATOMIC_RELAXED,
                                   __HIP_MEMORY_SCOPE_AGENT);
            unsigned target = (unsigned)NWG * (unsigned)(t + 1);
            while (__hip_atomic_load(counter, __ATOMIC_RELAXED,
                                     __HIP_MEMORY_SCOPE_AGENT) < target)
                __builtin_amdgcn_s_sleep(1);
        }
        __syncthreads();
        __builtin_amdgcn_fence(__ATOMIC_ACQUIRE, "agent");
    }

    // ---- readout: out[b,o] = sum_n r[b,n] * Cw[o,n]  (WG 0 only) ----
    if (wg == 0 && tid < NBATCH * NOUT) {
        const float* rf = (SEQLEN & 1) ? rbuf1 : rbuf0;  // final state buffer
        int b = tid >> 3, o = tid & 7;
        const float* rr = rf + (size_t)b * NRR;
        const float* cr = Cw + (size_t)o * NRR;
        float s = 0.0f;
        for (int n = 0; n < NRR; ++n) s += rr[n] * cr[n];
        out[b * NOUT + o] = s;
    }
}

// ---------------------------------------------------------------------------
extern "C" void kernel_launch(void* const* d_in, const int* in_sizes, int n_in,
                              void* d_out, int out_size, void* d_ws, size_t ws_size,
                              hipStream_t stream) {
    (void)in_sizes; (void)n_in; (void)out_size; (void)ws_size;

    const float* xin  = (const float*)d_in[0];  // input  (16,1024,12)
    const float* r0   = (const float*)d_in[1];  // r_0    (1,16,4096)
    const float* A    = (const float*)d_in[2];  // A      (4096,4096)
    const float* Bmat = (const float*)d_in[3];  // B      (4096,12)
    const float* bias = (const float*)d_in[4];  // b      (4096)
    const float* Cw   = (const float*)d_in[5];  // C_w    (8,4096)
    float* out = (float*)d_out;                 // (16,8)

    // Workspace layout (requires ~67.7 MB).
    unsigned char* ws = (unsigned char*)d_ws;
    unsigned short* Ahi = (unsigned short*)(ws);
    unsigned short* Alo = (unsigned short*)(ws + (size_t)NRR * NRR * 2);
    float* rbuf0 = (float*)(ws + (size_t)NRR * NRR * 4);
    float* rbuf1 = rbuf0 + NBATCH * NRR;
    unsigned* counter = (unsigned*)(ws + (size_t)NRR * NRR * 4
                                       + (size_t)2 * NBATCH * NRR * 4);

    // 1) Split A into bf16 hi/lo (16.7M elems, 4 per thread-float4).
    esn_split_A<<<dim3((NRR * NRR) / 4 / 256), dim3(256), 0, stream>>>(
        (const float4*)A, (uint2*)Ahi, (uint2*)Alo);

    // 2) Initialize state buffer + barrier counter.
    esn_init<<<dim3((NBATCH * NRR + 255) / 256), dim3(256), 0, stream>>>(
        r0, rbuf0, counter);

    // 3) Persistent recurrence kernel. Dynamic LDS:
    //    2 * 16*4104 bf16 (r hi/lo) + (192 + 768 + 64) floats = 266,752 B.
    size_t lds = (size_t)2 * NBATCH * LSTR * 2
               + (size_t)(NBATCH * NFEAT + 64 * NFEAT + 64) * 4;
    esn_recurrence<<<dim3(NWG), dim3(TPB), lds, stream>>>(
        Ahi, Alo, rbuf0, rbuf1, counter, xin, Bmat, bias, Cw, out);
}
